// TimeAwareGNN_66675072303278
// MI455X (gfx1250) — compile-verified, matrix operand order checked by reading, not run on hardware
//
#include <hip/hip_runtime.h>

typedef __attribute__((ext_vector_type(16))) _Float16 v16h;
typedef __attribute__((ext_vector_type(8)))  float    v8f;

#define L2_WAVES 4

// ---------------- scratch zeroing ----------------
__global__ void zero_kernel(float* __restrict__ p, int n) {
    int i = blockIdx.x * blockDim.x + threadIdx.x;
    if (i < n) p[i] = 0.0f;
}

// ---- pack weights into WMMA-B f16 layouts.
// B (32x16 f16) layout: lane = (K-half)*16 + col; element j of v16h:
//   K = (j<8 ? 0 : 16) + (lane>>4)*8 + (j&7), col = lane&15.
// idx < 8192:        m2_w2 [32,256] -> w2h [16 n-tiles][32 lanes][16 halves]
// idx in [8192,8704): m1_w2 [16,16] -> w1h [32 lanes][16 halves], K>=16 zero-padded
__global__ void pack_weights_kernel(const float* __restrict__ w2, const float* __restrict__ m1w2,
                                    _Float16* __restrict__ w2h, _Float16* __restrict__ w1h) {
    int idx = blockIdx.x * blockDim.x + threadIdx.x;
    if (idx < 16 * 32 * 16) {
        int j    = idx & 15;
        int lane = (idx >> 4) & 31;
        int n    = idx >> 9;
        int kh   = lane >> 4;
        int col  = lane & 15;
        int K    = ((j < 8) ? 0 : 16) + kh * 8 + (j & 7);
        w2h[idx] = (_Float16)w2[K * 256 + n * 16 + col];
    } else if (idx < 16 * 32 * 16 + 32 * 16) {
        int t    = idx - 16 * 32 * 16;
        int j    = t & 15;
        int lane = t >> 4;
        int kh   = lane >> 4;
        int col  = lane & 15;
        int K    = ((j < 8) ? 0 : 16) + kh * 8 + (j & 7);
        w1h[t]   = (K < 16) ? (_Float16)m1w2[K * 16 + col] : (_Float16)0.0f;
    }
}

// ---------------- layer 1: 16-edge tiles via one V_WMMA_F32_16X16X32_F16 ----------------
// A (16e x 32 f16) = relu(attr*w1+b1) for K<16, zero for K>=16.  B = zero-padded m1_w2.
// D[e][o] = (h_e @ m1_w2)[o];  msg[e][o] = x_src[e] * (D[e][o] + b2[o]).
__global__ __launch_bounds__(L2_WAVES * 32) void l1_edge_wmma_kernel(
    const float* __restrict__ x, const int* __restrict__ src, const int* __restrict__ dst,
    const float* __restrict__ attr,
    const float* __restrict__ w1, const float* __restrict__ b1,   // m1_w1[16], m1_b1[16]
    const _Float16* __restrict__ w1h,                              // packed [32][16] f16
    const float* __restrict__ b2,                                  // m1_b2[16]
    float* __restrict__ summed1, float* __restrict__ cnt, int E) {
    __shared__ float xsl[L2_WAVES][16];
    __shared__ int   dl[L2_WAVES][16];

    int wave = threadIdx.x >> 5;
    int lane = threadIdx.x & 31;
    int tile = blockIdx.x * L2_WAVES + wave;
    int e0 = tile * 16;
    if (e0 >= E) return;                       // wave-uniform exit

    int el = lane & 15;
    int kh = lane >> 4;
    int eg = e0 + el;
    bool full = (e0 + 16 <= E);                // wave-uniform
    bool valid = eg < E;
    int ec = valid ? eg : (E - 1);

    float a = valid ? attr[ec] : 0.0f;
    v16h A;
#pragma unroll
    for (int j = 0; j < 16; ++j) {
        int K = ((j < 8) ? 0 : 16) + kh * 8 + (j & 7);
        float v = 0.0f;
        if (K < 16) {
            v = fmaf(a, w1[K], b1[K]);
            v = v > 0.0f ? v : 0.0f;
        }
        A[j] = (_Float16)v;
    }

    if (kh == 0) {
        xsl[wave][el] = x[src[ec]];
        dl[wave][el]  = dst[ec];
    }
    v16h B = *(const v16h*)(w1h + (lane << 4));
    __syncthreads();

    v8f c = {};
    c = __builtin_amdgcn_wmma_f32_16x16x32_f16(
            false, A, false, B, (short)0, c, false, false);

    int col = lane & 15;
    float b2v = b2[col];
    if (full) {                                 // fast path: no per-row predication
#pragma unroll
        for (int r = 0; r < 8; ++r) {
            int M = r + kh * 8;
            float msg = xsl[wave][M] * (c[r] + b2v);
            atomicAdd(&summed1[(size_t)dl[wave][M] * 16 + col], msg);
        }
        if (kh == 0) atomicAdd(&cnt[dl[wave][el]], 1.0f);
    } else {
#pragma unroll
        for (int r = 0; r < 8; ++r) {
            int M = r + kh * 8;
            if (e0 + M < E) {
                float msg = xsl[wave][M] * (c[r] + b2v);
                atomicAdd(&summed1[(size_t)dl[wave][M] * 16 + col], msg);
            }
        }
        if (kh == 0 && valid) atomicAdd(&cnt[dl[wave][el]], 1.0f);
    }
}

// ---------------- finalize layer 1: mean + root + bias + relu ----------------
__global__ void fin1_kernel(const float* __restrict__ summed1, const float* __restrict__ cnt,
                            const float* __restrict__ x,
                            const float* __restrict__ root1, const float* __restrict__ bias1,
                            float* __restrict__ h1, int N) {
    int i = blockIdx.x * blockDim.x + threadIdx.x;
    if (i >= N * 16) return;
    int n = i >> 4, o = i & 15;
    float c = cnt[n];
    float agg = (c > 0.0f) ? summed1[i] / c : 0.0f;
    float v = fmaf(x[n], root1[o], agg + bias1[o]);
    h1[i] = v > 0.0f ? v : 0.0f;
}

// ---------------- layer 2: 16-edge tiles via 16x V_WMMA_F32_16X16X32_F16 ----------------
// A (16e x 32k f16) = relu(attr*w1+b1).  B tile n = w2[:, n*16:(n+1)*16] pre-packed.
// Output tile n == input channel i=n of W_e, so
//   msg[e][o] = sum_n x_src[e][n] * (D_n[e][o] + b2[n*16+o])  -> 8 FMAs/lane per WMMA.
__global__ __launch_bounds__(L2_WAVES * 32) void l2_edge_wmma_kernel(
    const int* __restrict__ src, const int* __restrict__ dst,
    const float* __restrict__ attr,
    const float* __restrict__ w1, const float* __restrict__ b1,      // m2_w1[32], m2_b1[32]
    const _Float16* __restrict__ w2h,                                // packed [16][32][16] f16
    const float* __restrict__ b2,                                    // m2_b2[256]
    const float* __restrict__ h1,
    float* __restrict__ summed2, int E) {
    __shared__ float xsl[L2_WAVES][16][16];
    __shared__ int   dl[L2_WAVES][16];

    int wave = threadIdx.x >> 5;
    int lane = threadIdx.x & 31;
    int tile = blockIdx.x * L2_WAVES + wave;
    int e0 = tile * 16;
    if (e0 >= E) return;                       // wave-uniform exit

    int el = lane & 15;
    int kh = lane >> 4;
    int eg = e0 + el;
    bool full = (e0 + 16 <= E);                // wave-uniform
    bool valid = eg < E;
    int ec = valid ? eg : (E - 1);

    float a = valid ? attr[ec] : 0.0f;
    v16h A;
#pragma unroll
    for (int j = 0; j < 16; ++j) {
        int K = ((j < 8) ? 0 : 16) + kh * 8 + (j & 7);
        float v = fmaf(a, w1[K], b1[K]);
        A[j] = (_Float16)(v > 0.0f ? v : 0.0f);
    }

    int s = src[ec];
#pragma unroll
    for (int c = 0; c < 8; ++c)
        xsl[wave][el][kh * 8 + c] = h1[(size_t)s * 16 + kh * 8 + c];
    if (kh == 0) dl[wave][el] = dst[ec];
    __syncthreads();

    int col = lane & 15;
    float b2v[16];
#pragma unroll
    for (int n = 0; n < 16; ++n) b2v[n] = b2[n * 16 + col];

    float acc[8];
#pragma unroll
    for (int r = 0; r < 8; ++r) acc[r] = 0.0f;

#pragma unroll
    for (int n = 0; n < 16; ++n) {
        v16h B = *(const v16h*)(w2h + ((n * 32 + lane) << 4));
        v8f c = {};
        c = __builtin_amdgcn_wmma_f32_16x16x32_f16(
                false, A, false, B, (short)0, c, false, false);
#pragma unroll
        for (int r = 0; r < 8; ++r) {
            int M = r + kh * 8;
            acc[r] = fmaf(xsl[wave][M][n], c[r] + b2v[n], acc[r]);
        }
    }

    if (full) {                                 // fast path: no per-row predication
#pragma unroll
        for (int r = 0; r < 8; ++r) {
            int M = r + kh * 8;
            atomicAdd(&summed2[(size_t)dl[wave][M] * 16 + col], acc[r]);
        }
    } else {
#pragma unroll
        for (int r = 0; r < 8; ++r) {
            int M = r + kh * 8;
            if (e0 + M < E)
                atomicAdd(&summed2[(size_t)dl[wave][M] * 16 + col], acc[r]);
        }
    }
}

// ---------------- finalize layer 2 + final Linear(16,1) ----------------
__global__ void fin2_kernel(const float* __restrict__ summed2, const float* __restrict__ cnt,
                            const float* __restrict__ h1,
                            const float* __restrict__ root2, const float* __restrict__ bias2,
                            const float* __restrict__ lin_w, const float* __restrict__ lin_b,
                            float* __restrict__ out, int N) {
    int n = blockIdx.x * blockDim.x + threadIdx.x;
    if (n >= N) return;
    float c = cnt[n];
    float inv = (c > 0.0f) ? 1.0f / c : 0.0f;
    float hr[16];
#pragma unroll
    for (int i = 0; i < 16; ++i) hr[i] = h1[(size_t)n * 16 + i];
    float res = lin_b[0];
#pragma unroll
    for (int o = 0; o < 16; ++o) {
        float v = fmaf(summed2[(size_t)n * 16 + o], inv, bias2[o]);
#pragma unroll
        for (int i = 0; i < 16; ++i) v = fmaf(hr[i], root2[i * 16 + o], v);
        v = v > 0.0f ? v : 0.0f;
        res = fmaf(v, lin_w[o], res);
    }
    out[n] = res;
}

extern "C" void kernel_launch(void* const* d_in, const int* in_sizes, int n_in,
                              void* d_out, int out_size, void* d_ws, size_t ws_size,
                              hipStream_t stream) {
    const float* x     = (const float*)d_in[0];
    const int*   eidx  = (const int*)d_in[1];   // [2,E] int32 (JAX x64 disabled)
    const float* attr  = (const float*)d_in[2];
    const float* m1_w1 = (const float*)d_in[3];
    const float* m1_b1 = (const float*)d_in[4];
    const float* m1_w2 = (const float*)d_in[5];
    const float* m1_b2 = (const float*)d_in[6];
    const float* root1 = (const float*)d_in[7];
    const float* bias1 = (const float*)d_in[8];
    const float* m2_w1 = (const float*)d_in[9];
    const float* m2_b1 = (const float*)d_in[10];
    const float* m2_w2 = (const float*)d_in[11];
    const float* m2_b2 = (const float*)d_in[12];
    const float* root2 = (const float*)d_in[13];
    const float* bias2 = (const float*)d_in[14];
    const float* lin_w = (const float*)d_in[15];
    const float* lin_b = (const float*)d_in[16];

    int N = in_sizes[0];        // x is [N,1]
    int E = in_sizes[2];        // edge_attr is [E,1]
    const int* src = eidx;
    const int* dst = eidx + E;

    // workspace (floats): summed1[N*16] | summed2[N*16] | cnt[N] | h1[N*16] | w2h,w1h (f16)
    float* ws      = (float*)d_ws;
    float* summed1 = ws;
    float* summed2 = ws + (size_t)N * 16;
    float* cnt     = ws + (size_t)N * 32;
    float* h1      = ws + (size_t)N * 33;
    size_t w2h_off = (((size_t)N * 49 * sizeof(float)) + 63) & ~(size_t)63;
    _Float16* w2h  = (_Float16*)((char*)d_ws + w2h_off);   // 16*32*16 halves
    _Float16* w1h  = w2h + 16 * 32 * 16;                   // 32*16 halves

    int zn = N * 33;  // summed1 + summed2 + cnt (contiguous)
    zero_kernel<<<(zn + 255) / 256, 256, 0, stream>>>(ws, zn);
    pack_weights_kernel<<<(16 * 32 * 16 + 32 * 16 + 255) / 256, 256, 0, stream>>>(
        m2_w2, m1_w2, w2h, w1h);

    int tiles = (E + 15) / 16;
    int tblocks = (tiles + L2_WAVES - 1) / L2_WAVES;

    l1_edge_wmma_kernel<<<tblocks, L2_WAVES * 32, 0, stream>>>(
        x, src, dst, attr, m1_w1, m1_b1, w1h, m1_b2, summed1, cnt, E);

    fin1_kernel<<<(N * 16 + 255) / 256, 256, 0, stream>>>(
        summed1, cnt, x, root1, bias1, h1, N);

    l2_edge_wmma_kernel<<<tblocks, L2_WAVES * 32, 0, stream>>>(
        src, dst, attr, m2_w1, m2_b1, w2h, m2_b2, h1, summed2, E);

    fin2_kernel<<<(N + 255) / 256, 256, 0, stream>>>(
        summed2, cnt, h1, root2, bias2, lin_w, lin_b, (float*)d_out, N);
}
